// Model_34325378629712
// MI455X (gfx1250) — matrix operand order repaired and checked by reading, not compiled, not run on hardware
//
#include <hip/hip_runtime.h>
#include <hip/hip_bf16.h>

typedef __attribute__((ext_vector_type(16))) _Float16 v16h;
typedef __attribute__((ext_vector_type(8)))  _Float16 v8h;
typedef __attribute__((ext_vector_type(8)))  float    v8f;

#define WAVES 8
#define INV_SQRT8   0.35355339059327373f
#define INV_SQRT32  0.17677669529663687f
#define INV_SQRT64  0.125f
#define INV_SQRT128 0.08838834764831845f

// Batched CDNA5 transcendental tanh: 8 independent v_tanh_f32 back-to-back
// (each is the "1 independent op" guard for the previous one), one trailing
// v_nop for the last per the TRANS hazard rule.
#define TANH8(x0,x1,x2,x3,x4,x5,x6,x7)                                        \
    asm("v_tanh_f32 %0, %0\n\tv_tanh_f32 %1, %1\n\tv_tanh_f32 %2, %2\n\t"     \
        "v_tanh_f32 %3, %3\n\tv_tanh_f32 %4, %4\n\tv_tanh_f32 %5, %5\n\t"     \
        "v_tanh_f32 %6, %6\n\tv_tanh_f32 %7, %7\n\tv_nop"                     \
        : "+v"(x0), "+v"(x1), "+v"(x2), "+v"(x3),                             \
          "+v"(x4), "+v"(x5), "+v"(x6), "+v"(x7))

__device__ inline v8f tanhact8(v8f x) {
    float t0=x[0], t1=x[1], t2=x[2], t3=x[3], t4=x[4], t5=x[5], t6=x[6], t7=x[7];
    TANH8(t0,t1,t2,t3,t4,t5,t6,t7);
    v8f r; r[0]=t0; r[1]=t1; r[2]=t2; r[3]=t3; r[4]=t4; r[5]=t5; r[6]=t6; r[7]=t7;
    return r;
}

// normact(x) = x * tanh(|x|) / (|x| + eps), elementwise on a v8f accumulator
__device__ inline v8f normact8(v8f x) {
    float a0=fabsf(x[0]), a1=fabsf(x[1]), a2=fabsf(x[2]), a3=fabsf(x[3]);
    float a4=fabsf(x[4]), a5=fabsf(x[5]), a6=fabsf(x[6]), a7=fabsf(x[7]);
    TANH8(a0,a1,a2,a3,a4,a5,a6,a7);
    v8f r;
    r[0] = x[0]*a0/(fabsf(x[0])+1e-8f); r[1] = x[1]*a1/(fabsf(x[1])+1e-8f);
    r[2] = x[2]*a2/(fabsf(x[2])+1e-8f); r[3] = x[3]*a3/(fabsf(x[3])+1e-8f);
    r[4] = x[4]*a4/(fabsf(x[4])+1e-8f); r[5] = x[5]*a5/(fabsf(x[5])+1e-8f);
    r[6] = x[6]*a6/(fabsf(x[6])+1e-8f); r[7] = x[7]*a7/(fabsf(x[7])+1e-8f);
    return r;
}

// Load a 32x16 (KxN) f16 B tile in WMMA B layout:
// lanes 0-15: K=k0..k0+7,k0+16..k0+23, N=n0+lane; lanes 16-31: K=k0+8..k0+15,k0+24..k0+31, N=n0+lane-16.
__device__ inline v16h load_B(const float* __restrict__ W, int ldn, int k0, int n0,
                              float scale, int kvalid) {
    int lane = threadIdx.x & 31;
    int n  = n0 + (lane & 15);
    int kb = k0 + ((lane >> 4) << 3);
    v16h b;
#pragma unroll
    for (int i = 0; i < 16; ++i) {
        int k = kb + i + ((i >> 3) << 3);
        float w = (k < kvalid) ? W[k * ldn + n] : 0.0f;
        b[i] = (_Float16)(w * scale);
    }
    return b;
}

// Load 16x32 f16 A tile (rows = edges/nodes) from LDS row-major, stride 72 halves.
__device__ inline v16h load_A(const _Float16* base, int ktile) {
    int lane = threadIdx.x & 31;
    int row = lane & 15;
    int kb  = (lane >> 4) * 8;
    const _Float16* p = base + row * 72 + ktile * 32 + kb;   // 16B aligned
    v8h lo = *(const v8h*)p;
    v8h hi = *(const v8h*)(p + 16);
    v16h a;
#pragma unroll
    for (int i = 0; i < 8; ++i) { a[i] = lo[i]; a[i + 8] = hi[i]; }
    return a;
}

// Store a WMMA D tile (v8f) as f16 into LDS row-major (stride 72 halves).
__device__ inline void store_D(v8f c, _Float16* base, int n0) {
    int lane = threadIdx.x & 31;
    int col = n0 + (lane & 15);
    int r0  = (lane >> 4) * 8;
#pragma unroll
    for (int rr = 0; rr < 8; ++rr)
        base[(r0 + rr) * 72 + col] = (_Float16)c[rr];
}

#define WMMA(a, b, c) __builtin_amdgcn_wmma_f32_16x16x32_f16(false, a, false, b, (short)0, c, false, false)

// ---------------- prep: zero agg, build species lookup tables ----------------
__global__ void prep_kernel(const float* __restrict__ Wna,   // [4,32]
                            const float* __restrict__ Wlin1, // [32,32]
                            const float* __restrict__ Wsc,   // [32,4,64]
                            float* __restrict__ agg, float* __restrict__ x1tab,
                            float* __restrict__ sctab, int aggCount) {
    int stride = gridDim.x * blockDim.x;
    for (int i = blockIdx.x * blockDim.x + threadIdx.x; i < aggCount; i += stride)
        agg[i] = 0.0f;
    if (blockIdx.x == 0) {
        int t = threadIdx.x;
        if (t < 128) {                       // x1[spec][f] (includes onehot/2 and 1/sqrt(32))
            int sp = t >> 5, f = t & 31;
            float acc = 0.0f;
#pragma unroll
            for (int i = 0; i < 32; ++i) acc += Wna[sp * 32 + i] * Wlin1[i * 32 + f];
            x1tab[t] = acc * 0.5f * INV_SQRT32;
        }
        if (t < 256) {                       // self-connection sc[spec][k]
            int sp = t >> 6, k = t & 63;
            float acc = 0.0f;
#pragma unroll
            for (int i = 0; i < 32; ++i) acc += Wna[sp * 32 + i] * Wsc[i * 256 + sp * 64 + k];
            sctab[t] = acc * 0.5f * INV_SQRT128;
        }
    }
}

// ---------------- edge kernel: radial MLP (WMMA) + message scatter ----------------
__global__ __launch_bounds__(256) void edge_kernel(
    const float* __restrict__ coords, const int* __restrict__ species,
    const int* __restrict__ edge_index,
    const float* __restrict__ Wr1, const float* __restrict__ Wr2,
    const float* __restrict__ Wr3,
    const float* __restrict__ x1tab, float* __restrict__ agg,
    int E, int ntile) {
    __shared__ __align__(16) _Float16 sH[WAVES][16][72];   // layer-1 activations
    __shared__ __align__(16) _Float16 sG[WAVES][16][72];   // layer-2 activations
    __shared__ int   sDst[WAVES][16];
    __shared__ int   sSpec[WAVES][16];
    __shared__ float sX1[128];

    const int lane = threadIdx.x & 31;
    const int wave = threadIdx.x >> 5;

    if (threadIdx.x < 128) sX1[threadIdx.x] = x1tab[threadIdx.x];

    // B tiles in registers for the whole kernel (normalizations folded in)
    v16h b1[4], b2[2][4], b3[2][2];
#pragma unroll
    for (int nt = 0; nt < 4; ++nt) b1[nt] = load_B(Wr1, 64, 0, nt * 16, INV_SQRT8, 8);
#pragma unroll
    for (int kt = 0; kt < 2; ++kt)
#pragma unroll
        for (int nt = 0; nt < 4; ++nt) b2[kt][nt] = load_B(Wr2, 64, kt * 32, nt * 16, INV_SQRT64, 64);
#pragma unroll
    for (int kt = 0; kt < 2; ++kt)
#pragma unroll
        for (int nt = 0; nt < 2; ++nt) b3[kt][nt] = load_B(Wr3, 64, kt * 32, nt * 16, INV_SQRT64, 64);
    __syncthreads();

    const int ngrp = (ntile + WAVES - 1) / WAVES;
    for (int g = blockIdx.x; g < ngrp; g += gridDim.x) {
        const int tile = g * WAVES + wave;
        const bool act = (tile < ntile);

        // ---- phase 1: per-edge radial features (lanes 0-15 own one edge each) ----
        float ea[8];
#pragma unroll
        for (int k = 0; k < 8; ++k) ea[k] = 0.0f;
        if (act && lane < 16) {
            int e = tile * 16 + lane;
            if (e < E) {
                int s = edge_index[e];
                int d = edge_index[E + e];
                float dx = coords[3 * d + 0] - coords[3 * s + 0];
                float dy = coords[3 * d + 1] - coords[3 * s + 1];
                float dz = coords[3 * d + 2] - coords[3 * s + 2];
                float r = sqrtf(dx * dx + dy * dy + dz * dz + 1e-8f);
                float u = r * 0.25f;                         // r / RMAX
                float fc = 0.0f;
                if (u < 1.0f) {
                    float u2 = u * u, u6 = u2 * u2 * u2;
                    fc = 1.0f + u6 * (-28.0f + u * 48.0f - u2 * 21.0f);
                }
                float pref = 0.5f * fc / r;                  // (2/RMAX)*fcut/r
                float base = r * 0.78539816339744831f;       // pi*r/RMAX
#pragma unroll
                for (int k = 0; k < 8; ++k) ea[k] = pref * __sinf(base * (float)(k + 1));
                sDst[wave][lane]  = d;
                sSpec[wave][lane] = species[s];
            } else {
                sDst[wave][lane] = -1;
                sSpec[wave][lane] = 0;
            }
        }
        __syncthreads();

        // ---- layer 1: [16,8(pad32)] x [32,64] via 4 WMMAs; A straight from regs ----
        if (act) {
            v16h a1;
#pragma unroll
            for (int i = 0; i < 8; ++i) { a1[i] = (_Float16)ea[i]; a1[i + 8] = (_Float16)0.0f; }
#pragma unroll
            for (int nt = 0; nt < 4; ++nt) {
                v8f c = {};
                c = WMMA(a1, b1[nt], c);
                store_D(tanhact8(c), &sH[wave][0][0], nt * 16);
            }
        }
        __syncthreads();

        // ---- layer 2: [16,64] x [64,64] via 8 WMMAs ----
        if (act) {
            v16h a0 = load_A(&sH[wave][0][0], 0);
            v16h a1k = load_A(&sH[wave][0][0], 1);
#pragma unroll
            for (int nt = 0; nt < 4; ++nt) {
                v8f c = {};
                c = WMMA(a0, b2[0][nt], c);
                c = WMMA(a1k, b2[1][nt], c);
                store_D(tanhact8(c), &sG[wave][0][0], nt * 16);
            }
        }
        __syncthreads();

        // ---- layer 3: [16,64] x [64,32] via 4 WMMAs, then message scatter ----
        if (act) {
            v16h a0 = load_A(&sG[wave][0][0], 0);
            v16h a1k = load_A(&sG[wave][0][0], 1);
#pragma unroll
            for (int nt = 0; nt < 2; ++nt) {
                v8f c = {};
                c = WMMA(a0, b3[0][nt], c);
                c = WMMA(a1k, b3[1][nt], c);
                int col = nt * 16 + (lane & 15);
                int r0  = (lane >> 4) * 8;
#pragma unroll
                for (int rr = 0; rr < 8; ++rr) {
                    int er = r0 + rr;
                    int dn = sDst[wave][er];
                    if (dn >= 0) {
                        int sp = sSpec[wave][er];
                        float msg = c[rr] * sX1[sp * 32 + col];   // w * x1[src] (0e path)
                        atomicAdd(&agg[(size_t)dn * 32 + col], msg);
                    }
                }
            }
        }
        __syncthreads();
    }
}

// ---------------- node kernel: post-aggregation MLP chain, all WMMA ----------------
__global__ __launch_bounds__(256) void node_kernel(
    const int* __restrict__ species, const float* __restrict__ agg,
    const float* __restrict__ sctab,
    const float* __restrict__ W2s, const float* __restrict__ Wfin,
    const float* __restrict__ Wr,  const float* __restrict__ Wm1,
    const float* __restrict__ Wm2, const float* __restrict__ Wm3,
    float* __restrict__ out, int NN, int ntile) {
    __shared__ __align__(16) _Float16 bufA[WAVES][16][72];  // 64-wide stages (s, m1)
    __shared__ __align__(16) _Float16 bufB[WAVES][16][72];  // 32-wide stages (y0, y, m2)
    __shared__ int   sSpec[WAVES][16];
    __shared__ float sSc[256];
    __shared__ float sWm3[32];

    const int lane = threadIdx.x & 31;
    const int wave = threadIdx.x >> 5;

    for (int i = threadIdx.x; i < 256; i += 256) sSc[i] = sctab[i];
    if (threadIdx.x < 32) sWm3[threadIdx.x] = Wm3[threadIdx.x];

    // weight tiles in registers, B layout, normalizations folded in
    v16h bW2s[4], bWfin[2][2], bWr[2], bWm1[4], bWm2[2][2];
#pragma unroll
    for (int nt = 0; nt < 4; ++nt) bW2s[nt] = load_B(W2s, 64, 0, nt * 16, 1.0f, 32);
#pragma unroll
    for (int kt = 0; kt < 2; ++kt)
#pragma unroll
        for (int nt = 0; nt < 2; ++nt) bWfin[kt][nt] = load_B(Wfin, 32, kt * 32, nt * 16, INV_SQRT64, 64);
#pragma unroll
    for (int nt = 0; nt < 2; ++nt) bWr[nt] = load_B(Wr, 32, 0, nt * 16, INV_SQRT32, 32);
#pragma unroll
    for (int nt = 0; nt < 4; ++nt) bWm1[nt] = load_B(Wm1, 64, 0, nt * 16, INV_SQRT32, 32);
#pragma unroll
    for (int kt = 0; kt < 2; ++kt)
#pragma unroll
        for (int nt = 0; nt < 2; ++nt) bWm2[kt][nt] = load_B(Wm2, 32, kt * 32, nt * 16, INV_SQRT64, 64);
    __syncthreads();

    const int ngrp = (ntile + WAVES - 1) / WAVES;
    for (int g = blockIdx.x; g < ngrp; g += gridDim.x) {
        const int tile = g * WAVES + wave;
        const bool act = (tile < ntile);

        // ---- P0: build A (16 nodes x 32 feats) in regs directly from agg ----
        v16h aIn;
        if (act) {
            int node = tile * 16 + (lane & 15);
            if (node >= NN) node = NN - 1;                    // clamp (dup load, store guarded)
            const float* arow = agg + (size_t)node * 32;
            const float sc = 0.0625f * INV_SQRT32;            // /AVG_NEI and W2_s scale
            int kb = (lane >> 4) * 8;
#pragma unroll
            for (int i = 0; i < 8; ++i) {
                aIn[i]     = (_Float16)(arow[kb + i] * sc);
                aIn[i + 8] = (_Float16)(arow[16 + kb + i] * sc);
            }
            if (lane < 16) sSpec[wave][lane] = species[node];
        }
        __syncthreads();

        // ---- P1: s = normact(a @ W2s + sc[spec])  [16,64] ----
        if (act) {
#pragma unroll
            for (int nt = 0; nt < 4; ++nt) {
                v8f c = {};
                c = WMMA(aIn, bW2s[nt], c);
                int col = nt * 16 + (lane & 15);
                int r0  = (lane >> 4) * 8;
#pragma unroll
                for (int rr = 0; rr < 8; ++rr)
                    c[rr] += sSc[sSpec[wave][r0 + rr] * 64 + col];
                store_D(normact8(c), &bufA[wave][0][0], nt * 16);
            }
        }
        __syncthreads();

        // ---- P2: y0 = normact(s @ Wfin) [16,32]; keep y0 in regs, stage to bufB ----
        v8f y0c[2];
        if (act) {
            v16h a0 = load_A(&bufA[wave][0][0], 0);
            v16h a1 = load_A(&bufA[wave][0][0], 1);
#pragma unroll
            for (int nt = 0; nt < 2; ++nt) {
                v8f c = {};
                c = WMMA(a0, bWfin[0][nt], c);
                c = WMMA(a1, bWfin[1][nt], c);
                y0c[nt] = normact8(c);
                store_D(y0c[nt], &bufB[wave][0][0], nt * 16);
            }
        }
        __syncthreads();

        // ---- P3: y = y0 + normact(y0 @ Wr) [16,32] ----
        if (act) {
            v16h a0 = load_A(&bufB[wave][0][0], 0);
#pragma unroll
            for (int nt = 0; nt < 2; ++nt) {
                v8f c = {};
                c = WMMA(a0, bWr[nt], c);
                v8f y = y0c[nt] + normact8(c);
                store_D(y, &bufB[wave][0][0], nt * 16);
            }
        }
        __syncthreads();

        // ---- P4: m1 = tanh(y @ Wm1) [16,64] ----
        if (act) {
            v16h a0 = load_A(&bufB[wave][0][0], 0);
#pragma unroll
            for (int nt = 0; nt < 4; ++nt) {
                v8f c = {};
                c = WMMA(a0, bWm1[nt], c);
                store_D(tanhact8(c), &bufA[wave][0][0], nt * 16);
            }
        }
        __syncthreads();

        // ---- P5: m2 = tanh(m1 @ Wm2) [16,32] ----
        if (act) {
            v16h a0 = load_A(&bufA[wave][0][0], 0);
            v16h a1 = load_A(&bufA[wave][0][0], 1);
#pragma unroll
            for (int nt = 0; nt < 2; ++nt) {
                v8f c = {};
                c = WMMA(a0, bWm2[0][nt], c);
                c = WMMA(a1, bWm2[1][nt], c);
                store_D(tanhact8(c), &bufB[wave][0][0], nt * 16);
            }
        }
        __syncthreads();

        // ---- P6: out = m2 @ Wm3 (lanes 0-15, one node each) ----
        if (act && lane < 16) {
            int node = tile * 16 + lane;
            if (node < NN) {
                float o = 0.0f;
#pragma unroll
                for (int j = 0; j < 32; ++j)
                    o += (float)bufB[wave][lane][j] * sWm3[j];
                out[node] = o * INV_SQRT32;
            }
        }
        __syncthreads();
    }
}

extern "C" void kernel_launch(void* const* d_in, const int* in_sizes, int n_in,
                              void* d_out, int out_size, void* d_ws, size_t ws_size,
                              hipStream_t stream) {
    const float* coords   = (const float*)d_in[0];
    const int*   species  = (const int*)d_in[1];
    const int*   edge_idx = (const int*)d_in[2];
    // d_in[3] contributions: unused by the reference forward
    const float* Wna   = (const float*)d_in[4];
    const float* Wlin1 = (const float*)d_in[5];
    const float* Wr1   = (const float*)d_in[6];
    const float* Wr2   = (const float*)d_in[7];
    const float* Wr3   = (const float*)d_in[8];
    const float* W2s   = (const float*)d_in[9];
    // d_in[10] W2_v: dead code (vector path dropped by final scalar-only linear)
    const float* Wsc   = (const float*)d_in[11];
    const float* Wfin  = (const float*)d_in[12];
    const float* Wr    = (const float*)d_in[13];
    const float* Wm1   = (const float*)d_in[14];
    const float* Wm2   = (const float*)d_in[15];
    const float* Wm3   = (const float*)d_in[16];
    float* out = (float*)d_out;

    const int N = in_sizes[1];
    const int E = in_sizes[2] / 2;

    float* agg   = (float*)d_ws;                 // [N,32]
    float* x1tab = agg + (size_t)N * 32;         // [4,32]
    float* sctab = x1tab + 128;                  // [4,64]

    prep_kernel<<<512, 256, 0, stream>>>(Wna, Wlin1, Wsc, agg, x1tab, sctab, N * 32);

    const int etile = (E + 15) / 16;
    const int egrp  = (etile + WAVES - 1) / WAVES;
    int eblocks = egrp < 2048 ? egrp : 2048;
    edge_kernel<<<eblocks, 256, 0, stream>>>(coords, species, edge_idx,
                                             Wr1, Wr2, Wr3, x1tab, agg, E, etile);

    const int vtile = (N + 15) / 16;
    const int vgrp  = (vtile + WAVES - 1) / WAVES;
    int vblocks = vgrp < 2048 ? vgrp : 2048;
    node_kernel<<<vblocks, 256, 0, stream>>>(species, agg, sctab,
                                             W2s, Wfin, Wr, Wm1, Wm2, Wm3,
                                             out, N, vtile);
}
